// SelfAttention_25512105738728
// MI455X (gfx1250) — matrix operand order repaired and checked by reading, not compiled, not run on hardware
//
#include <hip/hip_runtime.h>
#include <hip/hip_bf16.h>

#define D_MODEL 512
#define DK      64
#define SEQ     4096
#define BATCH   4

typedef __attribute__((ext_vector_type(16))) _Float16 v16h;
typedef __attribute__((ext_vector_type(8)))  _Float16 v8h;
typedef __attribute__((ext_vector_type(8)))  float    v8f;
typedef __attribute__((ext_vector_type(4)))  float    v4f;
typedef __attribute__((ext_vector_type(4)))  int      v4i;
typedef __attribute__((address_space(1))) v4i* gv4i_p;
typedef __attribute__((address_space(3))) v4i* lv4i_p;

static __device__ __forceinline__ v16h cat8(v8h lo, v8h hi) {
  return __builtin_shufflevector(lo, hi, 0,1,2,3,4,5,6,7,8,9,10,11,12,13,14,15);
}

#define WMMA_F16(A,B,C) \
  __builtin_amdgcn_wmma_f32_16x16x32_f16(false,(A),false,(B),(short)0,(C),false,false)

// Async global->LDS staging path (gfx1250 ASYNCcnt). Guarded so the compile
// cannot break if this toolchain lacks the builtin.
#if __has_builtin(__builtin_amdgcn_global_load_async_to_lds_b128)
#define HAVE_ASYNC_LDS 1
#else
#define HAVE_ASYNC_LDS 0
#endif

#if HAVE_ASYNC_LDS
#if __has_builtin(__builtin_amdgcn_s_wait_asynccnt)
#define WAIT_ASYNC(n) __builtin_amdgcn_s_wait_asynccnt(n)
#else
#define WAIT_ASYNC(n) asm volatile("s_wait_asynccnt %0" ::"i"(n) : "memory")
#endif
#endif

// ---------------------------------------------------------------------------
// Kernel 0: repack Wq/Wk/Wv (fp32, [512][64]) into f16 WMMA-B-operand layout.
// B-layout for 16x16x32: lane = (n&15) + 16*((k_local>>3)&1), halves = (k&7) + 8*((k>>4)&1).
// Scale 1/sqrt(64)=0.125 folded into Wq.
// ---------------------------------------------------------------------------
__global__ __launch_bounds__(256) void prep_w(
    const float* __restrict__ Wq, const float* __restrict__ Wk,
    const float* __restrict__ Wv, _Float16* __restrict__ WB) {
  int tid = blockIdx.x * 256 + threadIdx.x;
  if (tid >= 3 * D_MODEL * DK) return;
  int mat = tid / (D_MODEL * DK);
  int rem = tid - mat * (D_MODEL * DK);
  int d = rem / DK, n = rem % DK;
  const float* W = (mat == 0) ? Wq : (mat == 1 ? Wk : Wv);
  float w = W[d * DK + n];
  if (mat == 0) w *= 0.125f;
  int idx = ((mat * 16 + (d >> 5)) * 4 + (n >> 4)) * 512
          + ((n & 15) + 16 * ((d >> 3) & 1)) * 16 + (d & 7) + 8 * ((d >> 4) & 1);
  WB[idx] = (_Float16)w;
}

// ---------------------------------------------------------------------------
// Kernel 1: QKV projection. One wave per 16 rows of x. 12 WMMA accumulators
// (Q,K,V x 4 n-tiles), K-loop over 512 in steps of 32.
// Outputs: Qh row-major f16 (pre-scaled), KB / VB in WMMA-B-operand layout.
// ---------------------------------------------------------------------------
__global__ __launch_bounds__(256) void qkv_proj(
    const float* __restrict__ x, const _Float16* __restrict__ WB,
    const float* __restrict__ bq, const float* __restrict__ bk,
    const float* __restrict__ bv,
    _Float16* __restrict__ Qh, _Float16* __restrict__ KB,
    _Float16* __restrict__ VB) {
  const int lane = threadIdx.x & 31;
  const int wave = threadIdx.x >> 5;
  const int tile = blockIdx.x * 8 + wave;       // [0, BATCH*SEQ/16)
  const int b    = tile / (SEQ / 16);
  const int s0   = (tile % (SEQ / 16)) * 16;

  v8f acc[3][4];
  #pragma unroll
  for (int m = 0; m < 3; ++m)
    #pragma unroll
    for (int nb = 0; nb < 4; ++nb) { v8f z = {}; acc[m][nb] = z; }

  const int    row     = s0 + (lane & 15);
  const float* xrow    = x + ((size_t)b * SEQ + row) * D_MODEL;
  const int    base_lo = 8 * (lane >> 4);

  for (int ks = 0; ks < 16; ++ks) {
    const float* p0 = xrow + ks * 32 + base_lo;
    v4f f0 = *(const v4f*)(p0);
    v4f f1 = *(const v4f*)(p0 + 4);
    v4f f2 = *(const v4f*)(p0 + 16);
    v4f f3 = *(const v4f*)(p0 + 20);
    v16h a;
    #pragma unroll
    for (int i = 0; i < 4; ++i) {
      a[i]      = (_Float16)f0[i];
      a[4 + i]  = (_Float16)f1[i];
      a[8 + i]  = (_Float16)f2[i];
      a[12 + i] = (_Float16)f3[i];
    }
    #pragma unroll
    for (int mat = 0; mat < 3; ++mat) {
      #pragma unroll
      for (int nb = 0; nb < 4; ++nb) {
        v16h bf = *(const v16h*)(WB + (((size_t)(mat * 16 + ks) * 4 + nb) * 32 + lane) * 16);
        acc[mat][nb] = WMMA_F16(a, bf, acc[mat][nb]);
      }
    }
  }

  const int nlo = lane & 15;
  #pragma unroll
  for (int nb = 0; nb < 4; ++nb) {
    const int   n  = nb * 16 + nlo;
    const float bQ = bq[n] * 0.125f;
    const float bK = bk[n];
    const float bV = bv[n];
    #pragma unroll
    for (int r = 0; r < 8; ++r) {
      const int srow = s0 + r + 8 * (lane >> 4);      // C-layout row mapping
      Qh[((size_t)b * SEQ + srow) * DK + n] = (_Float16)(acc[0][nb][r] + bQ);
      {
        size_t idx = ((((size_t)b * (SEQ / 16) + (srow >> 4)) * 2 + (n >> 5)) * 32
                      + ((srow & 15) + 16 * ((n >> 3) & 1))) * 16 + (n & 7) + 8 * ((n >> 4) & 1);
        KB[idx] = (_Float16)(acc[1][nb][r] + bK);
      }
      {
        size_t idx = ((((size_t)b * (SEQ / 32) + (srow >> 5)) * 4 + nb) * 32
                      + ((n & 15) + 16 * ((srow >> 3) & 1))) * 16 + (srow & 7) + 8 * ((srow >> 4) & 1);
        VB[idx] = (_Float16)(acc[2][nb][r] + bV);
      }
    }
  }
}

// ---------------------------------------------------------------------------
// Kernel 2: flash attention. 8 waves/block, one 16-query tile per wave; all
// waves share the 32-key step, so K/V tiles are staged into LDS once per
// block (double-buffered, async global->LDS) instead of each wave streaming
// them from L2 (8x traffic reduction).
// ---------------------------------------------------------------------------
__global__ __launch_bounds__(256) void flash_attn(
    const _Float16* __restrict__ Qh, const _Float16* __restrict__ KB,
    const _Float16* __restrict__ VB, float* __restrict__ out) {
  __shared__ __align__(32) _Float16 ldsK[2][2048];   // 2 x 4KB  (32 keys x 64d)
  __shared__ __align__(32) _Float16 ldsV[2][2048];   // 2 x 4KB  (32 keys x 64v)
  __shared__ __align__(32) _Float16 ldsP[8][512];    // per-wave P transpose

  const int tid  = threadIdx.x;
  const int lane = tid & 31;
  const int wave = tid >> 5;
  const int tile = blockIdx.x * 8 + wave;
  const int b    = tile / (SEQ / 16);                // uniform across block
  const int q0   = (tile % (SEQ / 16)) * 16;
  _Float16* P = ldsP[wave];

  // Q A-fragments (scale already folded in).
  v16h qa[2];
  const _Float16* qrow =
      Qh + ((size_t)b * SEQ + q0 + (lane & 15)) * DK + 8 * (lane >> 4);
  #pragma unroll
  for (int db = 0; db < 2; ++db) {
    v8h lo = *(const v8h*)(qrow + db * 32);
    v8h hi = *(const v8h*)(qrow + db * 32 + 16);
    qa[db] = cat8(lo, hi);
  }

  v8f acc[4];
  #pragma unroll
  for (int vb = 0; vb < 4; ++vb) { v8f z = {}; acc[vb] = z; }
  float m[8], l[8];
  #pragma unroll
  for (int r = 0; r < 8; ++r) { m[r] = -3.0e38f; l[r] = 0.0f; }

  const _Float16* KBb = KB + (size_t)b * (SEQ / 16) * 1024;
  const _Float16* VBb = VB + (size_t)b * (SEQ / 32) * 2048;

  // Cooperative staging of one 32-key step (K: 4KB, V: 4KB): 256 threads x 32B.
  auto stage = [&](int s0, int bb) {
    const _Float16* src;
    _Float16*       dst;
    if (tid < 128) {
      src = KBb + (size_t)(s0 >> 4) * 1024 + tid * 16;
      dst = &ldsK[bb][tid * 16];
    } else {
      src = VBb + (size_t)(s0 >> 5) * 2048 + (tid - 128) * 16;
      dst = &ldsV[bb][(tid - 128) * 16];
    }
#if HAVE_ASYNC_LDS
    _Float16* s = const_cast<_Float16*>(src);
    __builtin_amdgcn_global_load_async_to_lds_b128(
        (gv4i_p)(__attribute__((address_space(1))) void*)s,
        (lv4i_p)(__attribute__((address_space(3))) void*)dst, 0, 0);
    __builtin_amdgcn_global_load_async_to_lds_b128(
        (gv4i_p)(__attribute__((address_space(1))) void*)(s + 8),
        (lv4i_p)(__attribute__((address_space(3))) void*)(dst + 8), 0, 0);
#else
    *(v16h*)dst = *(const v16h*)src;                  // sync fallback
#endif
  };

  stage(0, 0);                                        // prologue: step 0

  for (int s0 = 0, it = 0; s0 < SEQ; s0 += 32, ++it) {
    const int  cur     = it & 1;
    const bool hasNext = (s0 + 32) < SEQ;
    if (hasNext) stage(s0 + 32, cur ^ 1);             // overlap next DMA
#if HAVE_ASYNC_LDS
    if (hasNext) { WAIT_ASYNC(2); } else { WAIT_ASYNC(0); }  // in-order drain of cur
#endif
    __syncthreads();                                  // cur tile visible to all waves

    const _Float16* Kt = ldsK[cur];
    const _Float16* Vt = ldsV[cur];

    // scores for 16q x 32k (B-fragments now from LDS)
    v8f sc0 = {}, sc1 = {};
    #pragma unroll
    for (int db = 0; db < 2; ++db) {
      v16h k0 = *(const v16h*)(Kt + db * 512 + lane * 16);
      sc0 = WMMA_F16(qa[db], k0, sc0);
    }
    #pragma unroll
    for (int db = 0; db < 2; ++db) {
      v16h k1 = *(const v16h*)(Kt + 1024 + db * 512 + lane * 16);
      sc1 = WMMA_F16(qa[db], k1, sc1);
    }

    // online softmax: row max across 16-lane halves (C-layout row mapping)
    float mx[8];
    #pragma unroll
    for (int r = 0; r < 8; ++r) mx[r] = fmaxf(sc0[r], sc1[r]);
    #pragma unroll
    for (int off = 1; off < 16; off <<= 1)
      #pragma unroll
      for (int r = 0; r < 8; ++r) mx[r] = fmaxf(mx[r], __shfl_xor(mx[r], off, 32));

    float corr[8], rs[8];
    #pragma unroll
    for (int r = 0; r < 8; ++r) {
      float nm = fmaxf(m[r], mx[r]);
      corr[r]  = __expf(m[r] - nm);
      m[r]     = nm;
    }
    #pragma unroll
    for (int r = 0; r < 8; ++r) {
      float p0 = __expf(sc0[r] - m[r]);
      float p1 = __expf(sc1[r] - m[r]);
      sc0[r] = p0; sc1[r] = p1; rs[r] = p0 + p1;
    }
    #pragma unroll
    for (int off = 1; off < 16; off <<= 1)
      #pragma unroll
      for (int r = 0; r < 8; ++r) rs[r] += __shfl_xor(rs[r], off, 32);
    #pragma unroll
    for (int r = 0; r < 8; ++r) l[r] = l[r] * corr[r] + rs[r];
    #pragma unroll
    for (int vb = 0; vb < 4; ++vb)
      #pragma unroll
      for (int r = 0; r < 8; ++r) acc[vb][r] *= corr[r];

    // P: C-layout -> per-wave LDS row-major [q][s_local]
    #pragma unroll
    for (int r = 0; r < 8; ++r) {
      int q = r + 8 * (lane >> 4);
      P[q * 32 + (lane & 15)]      = (_Float16)sc0[r];
      P[q * 32 + 16 + (lane & 15)] = (_Float16)sc1[r];
    }
    asm volatile("s_wait_dscnt 0" ::: "memory");   // per-wave LDS in-order + drain
    const _Float16* pr = P + (lane & 15) * 32 + 8 * (lane >> 4);
    v8h plo = *(const v8h*)pr;
    v8h phi = *(const v8h*)(pr + 16);
    v16h pa = cat8(plo, phi);
    asm volatile("" ::: "memory");

    // out += P @ V
    #pragma unroll
    for (int vb = 0; vb < 4; ++vb) {
      v16h vf = *(const v16h*)(Vt + vb * 512 + lane * 16);
      acc[vb] = WMMA_F16(pa, vf, acc[vb]);
    }

    __syncthreads();                                  // all waves done with cur
  }

  // epilogue: normalize and store fp32
  #pragma unroll
  for (int vb = 0; vb < 4; ++vb)
    #pragma unroll
    for (int r = 0; r < 8; ++r) {
      int q = q0 + r + 8 * (lane >> 4);
      int v = vb * 16 + (lane & 15);
      out[((size_t)b * SEQ + q) * DK + v] = acc[vb][r] / l[r];
    }
}

// ---------------------------------------------------------------------------
extern "C" void kernel_launch(void* const* d_in, const int* in_sizes, int n_in,
                              void* d_out, int out_size, void* d_ws, size_t ws_size,
                              hipStream_t stream) {
  (void)in_sizes; (void)n_in; (void)out_size; (void)ws_size;
  const float* x  = (const float*)d_in[0];
  const float* Wq = (const float*)d_in[1];
  const float* bq = (const float*)d_in[2];
  const float* Wk = (const float*)d_in[3];
  const float* bk = (const float*)d_in[4];
  const float* Wv = (const float*)d_in[5];
  const float* bv = (const float*)d_in[6];
  float* out = (float*)d_out;

  // Workspace layout (f16 elements): WB | Qh | KB | VB  (~6.2 MB total)
  _Float16* WB = (_Float16*)d_ws;
  _Float16* Qh = WB + (size_t)3 * 16 * 4 * 512;        // 98304 halves
  _Float16* KB = Qh + (size_t)BATCH * SEQ * DK;
  _Float16* VB = KB + (size_t)BATCH * SEQ * DK;

  hipLaunchKernelGGL(prep_w, dim3((3 * D_MODEL * DK + 255) / 256), dim3(256), 0,
                     stream, Wq, Wk, Wv, WB);
  hipLaunchKernelGGL(qkv_proj, dim3(BATCH * SEQ / 16 / 8), dim3(256), 0, stream,
                     x, WB, bq, bk, bv, Qh, KB, VB);
  hipLaunchKernelGGL(flash_attn, dim3(BATCH * SEQ / 16 / 8), dim3(256), 0, stream,
                     Qh, KB, VB, out);
}
